// GatedDeltaNet_58119497449728
// MI455X (gfx1250) — matrix operand order repaired and checked by reading, not compiled, not run on hardware
//
#include <hip/hip_runtime.h>

typedef __attribute__((ext_vector_type(16))) __bf16 v16bf;
typedef __attribute__((ext_vector_type(8)))  float  v8f;

#define BQ     2
#define NSEQ   2048
#define DMODEL 2048
#define NH     16
#define HD     128
#define INNER  2048
#define QKVC   6144
#define NTOK   (BQ * NSEQ)   // 4096 tokens

// ---------------------------------------------------------------- helpers
static __device__ __forceinline__ __bf16 f2bf(float f) {
  unsigned int u = __builtin_bit_cast(unsigned int, f);
  unsigned int r = u + 0x7fffu + ((u >> 16) & 1u);   // round-to-nearest-even
  unsigned short hs = (unsigned short)(r >> 16);
  return __builtin_bit_cast(__bf16, hs);
}

// low 32 bits of a generic LDS pointer == wave-relative LDS byte offset
static __device__ __forceinline__ unsigned ldsoff(const void* p) {
  return (unsigned)(unsigned long long)(uintptr_t)p;
}

// CDNA5 async copy engine: global -> LDS, 16B per lane, tracked by ASYNCcnt
static __device__ __forceinline__ void async_copy16(unsigned lds_dst, const void* gsrc) {
  asm volatile("global_load_async_to_lds_b128 %0, %1, off"
               :
               : "v"(lds_dst), "v"((unsigned long long)(uintptr_t)gsrc)
               : "memory");
}

static __device__ __forceinline__ void wait_async0() {
#if __has_builtin(__builtin_amdgcn_s_wait_asynccnt)
  __builtin_amdgcn_s_wait_asynccnt(0);
#else
  asm volatile("s_wait_asynccnt 0x0" ::: "memory");
#endif
}

__global__ __launch_bounds__(256) void cvt_bf16_kernel(const float* __restrict__ in,
                                                       __bf16* __restrict__ out, long n) {
  long i = (long)blockIdx.x * 256 + threadIdx.x;
  if (i < n) out[i] = f2bf(in[i]);
}

// ---------------------------------------------------------------- bf16 WMMA GEMM
// C[M,N] = A[M,K] * B[N,K]^T   (both row-major over K; M,N multiples of 128, K of 32)
union Frag16 {
  v16bf v;
  uint4 q[2];
};

__global__ __launch_bounds__(256) void gemm_bf16_wmma(const __bf16* __restrict__ A,
                                                      const __bf16* __restrict__ B,
                                                      float* __restrict__ C,
                                                      int M, int N, int K) {
  // double-buffered 128x32 bf16 tiles (8 KB each), 16B aligned for B128 async DMA
  __shared__ __align__(16) __bf16 As[2][128 * 32];
  __shared__ __align__(16) __bf16 Bs[2][128 * 32];

  const int tid  = threadIdx.x;
  const int lane = tid & 31;
  const int wv   = tid >> 5;     // 0..7
  const int wm   = wv >> 1;      // 0..3 : 32-row strip
  const int wn   = wv & 1;       // 0..1 : 64-col strip
  const int bm   = blockIdx.y * 128;
  const int bn   = blockIdx.x * 128;
  const int hi   = lane >> 4;    // lane half
  const int lr   = lane & 15;

  // per-thread tile-copy addressing: 2 chunks of 16B per matrix per K-step
  const int crow = tid >> 2;               // 0..63
  const int c8   = (tid & 3) * 8;          // K sub-offset in elements
  const __bf16* ga = A + (size_t)(bm + crow) * K + c8;
  const __bf16* gb = B + (size_t)(bn + crow) * K + c8;
  const size_t rowskip = (size_t)64 * K;   // +64 rows, in elements
  unsigned ldsA[2], ldsB[2];
  ldsA[0] = ldsoff(&As[0][0]) + tid * 16;
  ldsA[1] = ldsoff(&As[1][0]) + tid * 16;
  ldsB[0] = ldsoff(&Bs[0][0]) + tid * 16;
  ldsB[1] = ldsoff(&Bs[1][0]) + tid * 16;

  v8f acc[2][4];
#pragma unroll
  for (int s = 0; s < 2; ++s)
#pragma unroll
    for (int t = 0; t < 4; ++t)
#pragma unroll
      for (int r = 0; r < 8; ++r) acc[s][t][r] = 0.0f;

  // prologue: stage tile 0 into buffer 0
  async_copy16(ldsA[0],        ga);
  async_copy16(ldsA[0] + 4096, ga + rowskip);
  async_copy16(ldsB[0],        gb);
  async_copy16(ldsB[0] + 4096, gb + rowskip);
  wait_async0();
  __syncthreads();

  int cur = 0;
  for (int k0 = 0; k0 < K; k0 += 32) {
    // issue next tile's DMA into the other buffer; it overlaps the WMMAs below
    if (k0 + 32 < K) {
      const int nx = cur ^ 1;
      async_copy16(ldsA[nx],        ga + (k0 + 32));
      async_copy16(ldsA[nx] + 4096, ga + rowskip + (k0 + 32));
      async_copy16(ldsB[nx],        gb + (k0 + 32));
      async_copy16(ldsB[nx] + 4096, gb + rowskip + (k0 + 32));
    }

    const __bf16* Ab = &As[cur][0];
    const __bf16* Bb = &Bs[cur][0];

    Frag16 af[2], bfr[4];
    // A fragment: 16x32 bf16, K interleave 0-7/16-23 (lanes 0-15) vs 8-15/24-31
#pragma unroll
    for (int s = 0; s < 2; ++s) {
      int m = wm * 32 + s * 16 + lr;
      af[s].q[0] = *(const uint4*)&Ab[m * 32 + hi * 8];
      af[s].q[1] = *(const uint4*)&Ab[m * 32 + 16 + hi * 8];
    }
    // B fragment: row striped across lanes; K interleave 0-7/16-23 (lanes 0-15) vs 8-15/24-31
#pragma unroll
    for (int t = 0; t < 4; ++t) {
      int n = wn * 64 + t * 16 + lr;
      bfr[t].q[0] = *(const uint4*)&Bb[n * 32 + hi * 8];
      bfr[t].q[1] = *(const uint4*)&Bb[n * 32 + 16 + hi * 8];
    }
#pragma unroll
    for (int s = 0; s < 2; ++s)
#pragma unroll
      for (int t = 0; t < 4; ++t)
        acc[s][t] = __builtin_amdgcn_wmma_f32_16x16x32_bf16(
            false, af[s].v, false, bfr[t].v, (short)0, acc[s][t], false, false);

    // next tile must have landed in LDS before anyone reads it after the barrier
    wait_async0();
    __syncthreads();
    cur ^= 1;
  }

  // C/D layout: VGPR r -> M = r + 8*hi, N = lane&15
#pragma unroll
  for (int s = 0; s < 2; ++s)
#pragma unroll
    for (int t = 0; t < 4; ++t)
#pragma unroll
      for (int r = 0; r < 8; ++r) {
        int m = bm + wm * 32 + s * 16 + hi * 8 + r;
        int n = bn + wn * 64 + t * 16 + lr;
        C[(size_t)m * N + n] = acc[s][t][r];
      }
}

// ---------------------------------------------------------------- conv + silu + heads + l2norm
__global__ __launch_bounds__(128) void conv_heads_kernel(const float* __restrict__ qkv,
                                                         const float* __restrict__ conv_w,
                                                         float* __restrict__ qh,
                                                         float* __restrict__ kh,
                                                         float* __restrict__ vh) {
  const int n = blockIdx.x, h = blockIdx.y, b = blockIdx.z;
  const int d = threadIdx.x;
  const int cq = h * HD + d;
  const int ck = INNER + cq;
  const int cv = 2 * INNER + cq;

  float vq = 0.f, vk = 0.f, vv = 0.f;
#pragma unroll
  for (int j = 0; j < 4; ++j) {
    int nn = n - 3 + j;
    if (nn >= 0) {
      const float* row = qkv + (size_t)(b * NSEQ + nn) * QKVC;
      vq += row[cq] * conv_w[cq * 4 + j];
      vk += row[ck] * conv_w[ck * 4 + j];
      vv += row[cv] * conv_w[cv * 4 + j];
    }
  }
  vq = vq / (1.f + expf(-vq));   // silu
  vk = vk / (1.f + expf(-vk));
  vv = vv / (1.f + expf(-vv));

  __shared__ float red[128];
  red[d] = vq * vq; __syncthreads();
  for (int st = 64; st > 0; st >>= 1) { if (d < st) red[d] += red[d + st]; __syncthreads(); }
  float invq = 1.f / fmaxf(sqrtf(red[0]), 1e-12f);
  __syncthreads();
  red[d] = vk * vk; __syncthreads();
  for (int st = 64; st > 0; st >>= 1) { if (d < st) red[d] += red[d + st]; __syncthreads(); }
  float invk = 1.f / fmaxf(sqrtf(red[0]), 1e-12f);

  size_t o = (((size_t)(b * NH + h)) * NSEQ + n) * HD + d;
  qh[o] = vq * invq;
  kh[o] = vk * invk;
  vh[o] = vv;
}

// ---------------------------------------------------------------- alpha/beta GEMV + sigmoid
__global__ __launch_bounds__(256) void ab_kernel(const float* __restrict__ x,
                                                 const float* __restrict__ W_ab,
                                                 const float* __restrict__ b_ab,
                                                 float* __restrict__ alpha,
                                                 float* __restrict__ beta) {
  const int m = blockIdx.x;                 // token index
  const int tid = threadIdx.x;
  const int e = tid >> 3;                   // 0..31 output
  const int s = tid & 7;
  const float* xr = x + (size_t)m * DMODEL;
  const float* wr = W_ab + (size_t)e * DMODEL;
  float acc = 0.f;
  for (int d = s; d < DMODEL; d += 8) acc += xr[d] * wr[d];
  __shared__ float part[256];
  part[tid] = acc; __syncthreads();
  if (s == 0) {
    float sum = b_ab[e];
#pragma unroll
    for (int j = 0; j < 8; ++j) sum += part[(e << 3) + j];
    float sig = 1.f / (1.f + expf(-sum));
    int b = m / NSEQ, n = m % NSEQ;
    if (e < NH) alpha[(size_t)(b * NH + e) * NSEQ + n] = sig;
    else        beta [(size_t)(b * NH + (e - NH)) * NSEQ + n] = sig;
  }
}

// ---------------------------------------------------------------- delta-rule scan (S rows in VGPRs)
__global__ __launch_bounds__(128) void scan_kernel(const float* __restrict__ qh,
                                                   const float* __restrict__ kh,
                                                   const float* __restrict__ vh,
                                                   const float* __restrict__ alpha,
                                                   const float* __restrict__ beta,
                                                   float* __restrict__ oh) {
  const int bh = blockIdx.x;                // 0..B*NH-1
  const int d  = threadIdx.x;               // S row owned by this thread
  const float* q = qh + (size_t)bh * NSEQ * HD;
  const float* k = kh + (size_t)bh * NSEQ * HD;
  const float* v = vh + (size_t)bh * NSEQ * HD;
  const float* al = alpha + (size_t)bh * NSEQ;
  const float* be = beta  + (size_t)bh * NSEQ;
  float* o = oh + (size_t)bh * NSEQ * HD;

  float S[HD];
#pragma unroll
  for (int e = 0; e < HD; ++e) S[e] = 0.f;

  __shared__ float ks[HD], qs[HD], vs[HD];
  for (int t = 0; t < NSEQ; ++t) {
    ks[d] = k[(size_t)t * HD + d];
    qs[d] = q[(size_t)t * HD + d];
    vs[d] = v[(size_t)t * HD + d];
    __syncthreads();
    float a = al[t], bt = be[t];
    float Sk = 0.f;
#pragma unroll
    for (int e = 0; e < HD; ++e) Sk += S[e] * ks[e];
    float u = bt * (vs[d] - a * Sk);
    float ot = 0.f;
#pragma unroll
    for (int e = 0; e < HD; ++e) {
      float sn = a * S[e] + u * ks[e];
      S[e] = sn;
      ot += sn * qs[e];
    }
    o[(size_t)t * HD + d] = ot;
    __syncthreads();
  }
}

// ---------------------------------------------------------------- gate + bf16 pack
__global__ __launch_bounds__(256) void gate_kernel(const float* __restrict__ oh,
                                                   const float* __restrict__ g,
                                                   __bf16* __restrict__ gated) {
  size_t i = (size_t)blockIdx.x * 256 + threadIdx.x;
  if (i < (size_t)NTOK * INNER) {
    int b = (int)(i / ((size_t)NSEQ * INNER));
    size_t rem = i - (size_t)b * NSEQ * INNER;
    int n = (int)(rem / INNER);
    int c = (int)(rem % INNER);
    int h = c >> 7, dd = c & 127;
    float ov = oh[(((size_t)(b * NH + h)) * NSEQ + n) * HD + dd];
    float gv = g[i];
    gated[i] = f2bf(ov * (gv / (1.f + expf(-gv))));
  }
}

// ---------------------------------------------------------------- launch
extern "C" void kernel_launch(void* const* d_in, const int* in_sizes, int n_in,
                              void* d_out, int out_size, void* d_ws, size_t ws_size,
                              hipStream_t stream) {
  (void)in_sizes; (void)n_in; (void)out_size; (void)ws_size;
  const float* x     = (const float*)d_in[0];
  const float* Wqkv  = (const float*)d_in[1];
  const float* convw = (const float*)d_in[2];
  const float* Wab   = (const float*)d_in[3];
  const float* bab   = (const float*)d_in[4];
  const float* Wg    = (const float*)d_in[5];
  const float* Wo    = (const float*)d_in[6];
  float* y = (float*)d_out;

  char* ws = (char*)d_ws;
  size_t off = 0;
  auto alloc = [&](size_t bytes) { size_t o = off; off += (bytes + 255) & ~(size_t)255; return o; };

  __bf16* xb    = (__bf16*)(ws + alloc((size_t)NTOK * DMODEL * 2));
  __bf16* wqkvb = (__bf16*)(ws + alloc((size_t)QKVC * DMODEL * 2));
  __bf16* wgb   = (__bf16*)(ws + alloc((size_t)INNER * DMODEL * 2));
  __bf16* wob   = (__bf16*)(ws + alloc((size_t)DMODEL * INNER * 2));
  float*  qkv   = (float*)(ws + alloc((size_t)NTOK * QKVC * 4));
  float*  g     = (float*)(ws + alloc((size_t)NTOK * INNER * 4));
  float*  qh    = (float*)(ws + alloc((size_t)BQ * NH * NSEQ * HD * 4));
  float*  kh    = (float*)(ws + alloc((size_t)BQ * NH * NSEQ * HD * 4));
  float*  vh    = (float*)(ws + alloc((size_t)BQ * NH * NSEQ * HD * 4));
  float*  alpha = (float*)(ws + alloc((size_t)BQ * NH * NSEQ * 4));
  float*  beta  = (float*)(ws + alloc((size_t)BQ * NH * NSEQ * 4));
  // scan output + gated bf16 alias into the qkv region (qkv is dead after conv)
  float*  oh    = qkv;
  __bf16* gated = (__bf16*)((char*)qkv + (size_t)BQ * NH * NSEQ * HD * 4);

  // fp32 -> bf16 conversions for the WMMA GEMM operands
  cvt_bf16_kernel<<<(NTOK * DMODEL + 255) / 256, 256, 0, stream>>>(x, xb, (long)NTOK * DMODEL);
  cvt_bf16_kernel<<<(QKVC * DMODEL + 255) / 256, 256, 0, stream>>>(Wqkv, wqkvb, (long)QKVC * DMODEL);
  cvt_bf16_kernel<<<(INNER * DMODEL + 255) / 256, 256, 0, stream>>>(Wg, wgb, (long)INNER * DMODEL);
  cvt_bf16_kernel<<<(DMODEL * INNER + 255) / 256, 256, 0, stream>>>(Wo, wob, (long)DMODEL * INNER);

  // qkv = x @ W_qkv^T ; g = x @ W_g^T
  gemm_bf16_wmma<<<dim3(QKVC / 128, NTOK / 128), 256, 0, stream>>>(xb, wqkvb, qkv, NTOK, QKVC, DMODEL);
  gemm_bf16_wmma<<<dim3(INNER / 128, NTOK / 128), 256, 0, stream>>>(xb, wgb, g, NTOK, INNER, DMODEL);

  // gates, conv+heads, scan
  ab_kernel<<<NTOK, 256, 0, stream>>>(x, Wab, bab, alpha, beta);
  conv_heads_kernel<<<dim3(NSEQ, NH, BQ), 128, 0, stream>>>(qkv, convw, qh, kh, vh);
  scan_kernel<<<BQ * NH, 128, 0, stream>>>(qh, kh, vh, alpha, beta, oh);

  // out * silu(g) -> bf16 ; y = gated @ W_o^T
  gate_kernel<<<(NTOK * INNER + 255) / 256, 256, 0, stream>>>(oh, g, gated);
  gemm_bf16_wmma<<<dim3(DMODEL / 128, NTOK / 128), 256, 0, stream>>>(gated, wob, y, NTOK, DMODEL, INNER);
}
